// EnergyCarrier_55679956025710
// MI455X (gfx1250) — matrix operand order repaired and checked by reading, not compiled, not run on hardware
//
#include <hip/hip_runtime.h>
#include <hip/hip_bf16.h>
#include <math.h>

// ---------------------------------------------------------------------------
// EnergyCarrier fused forward for MI455X (gfx1250), wave32 + WMMA bf16.
// All GEMM operands pre-converted to bf16; inner loops are pure b128+WMMA.
// ---------------------------------------------------------------------------

typedef __attribute__((ext_vector_type(16))) __bf16        v16bf;
typedef __attribute__((ext_vector_type(8)))  float         v8f;
typedef __attribute__((ext_vector_type(8)))  unsigned short us8;

#define B_N  32768
#define H_D  1024
#define TH_D 3072   // 3*H

// fp32 -> bf16 with round-to-nearest-even
__device__ __forceinline__ unsigned short f2bf(float f) {
  union { float f; unsigned u; } c; c.f = f;
  unsigned u = c.u;
  u = (u + 0x7FFFu + ((u >> 16) & 1u)) >> 16;
  return (unsigned short)u;
}

union Frag { v16bf v; us8 h[2]; };

// A fragment (16x32 MxK bf16) from bf16 row-major [rows, ldk].
// ISA 16-bit A layout: lanes 0-15 (row m): K=k0+0..7, K=k0+16..23;
// lanes 16-31 (row m): K=k0+8..15, K=k0+24..31.  -> two 16B loads at +0, +16.
__device__ __forceinline__ v16bf load_a_bf16(const unsigned short* __restrict__ a,
                                             int m0, int ldk, int k0, int lane) {
  const unsigned short* p = a + (size_t)(m0 + (lane & 15)) * ldk + k0 + ((lane >> 4) << 3);
  Frag f;
  f.h[0] = *(const us8*)(p);
  f.h[1] = *(const us8*)(p + 16);
  return f.v;
}

// B fragment (32x16 KxN bf16). Column n of B = row n of weight W[N, ldk] (bf16).
// 16-bit B layout (matches A): lanes 0-15: K=k0+0..7, k0+16..23; lanes 16-31: K=k0+8..15, k0+24..31.
__device__ __forceinline__ v16bf load_b_bf16(const unsigned short* __restrict__ w,
                                             int n0, int ldk, int k0, int lane) {
  const unsigned short* p = w + (size_t)(n0 + (lane & 15)) * ldk + k0 + ((lane >> 4) << 3);
  Frag f;
  f.h[0] = *(const us8*)(p);
  f.h[1] = *(const us8*)(p + 16);
  return f.v;
}

#define WMMA_BF16(A, Bm, C) \
  __builtin_amdgcn_wmma_f32_16x16x32_bf16(false, (A), false, (Bm), (short)0, (C), false, false)

// ---------------------------------------------------------------------------
// Fused GRU cell: each wave computes a 32x16 tile of h_new (2 M-subtiles so
// every weight fragment feeds two WMMAs), carrying 12 f32 WMMA accumulators,
// then applies the gate math in registers. No [B,3H] intermediates.
// ---------------------------------------------------------------------------
__global__ __launch_bounds__(256)
void gru_layer_kernel(const unsigned short* __restrict__ Aih, int Kih, // [B,Kih] bf16
                      const unsigned short* __restrict__ Wih,   // [3H, Kih] bf16
                      const float* __restrict__ bih,            // [3H]
                      const unsigned short* __restrict__ Ahh,   // [B, H] bf16 (prev h)
                      const unsigned short* __restrict__ Whh,   // [3H, H] bf16
                      const float* __restrict__ bhh,            // [3H]
                      const float* __restrict__ Hprev,          // [B, H] fp32 (prev h)
                      float* __restrict__ Hout,                 // [B, H] fp32
                      unsigned short* __restrict__ HoutB)       // [B, H] bf16
{
  const int lane = threadIdx.x & 31;
  const int wave = threadIdx.x >> 5;
  const int tile = blockIdx.x * 8 + wave;
  const int nblk = tile & 63;        // H/16 = 64 column tiles
  const int mblk = tile >> 6;        // B/32 row tiles
  const int m0 = mblk * 32;
  const int n0 = nblk * 16;

  v8f z8 = {0,0,0,0,0,0,0,0};
  v8f ir0 = z8, iz0 = z8, in0 = z8, hr0 = z8, hz0 = z8, hn0 = z8;
  v8f ir1 = z8, iz1 = z8, in1 = z8, hr1 = z8, hz1 = z8, hn1 = z8;

  // input-to-hidden gates: K = Kih (64 padded for layer0, 1024 for layer1)
  for (int k0 = 0; k0 < Kih; k0 += 32) {
    v16bf a0 = load_a_bf16(Aih, m0,      Kih, k0, lane);
    v16bf a1 = load_a_bf16(Aih, m0 + 16, Kih, k0, lane);
    v16bf b0 = load_b_bf16(Wih, n0,           Kih, k0, lane);
    v16bf b1 = load_b_bf16(Wih, H_D + n0,     Kih, k0, lane);
    v16bf b2 = load_b_bf16(Wih, 2 * H_D + n0, Kih, k0, lane);
    ir0 = WMMA_BF16(a0, b0, ir0);  ir1 = WMMA_BF16(a1, b0, ir1);
    iz0 = WMMA_BF16(a0, b1, iz0);  iz1 = WMMA_BF16(a1, b1, iz1);
    in0 = WMMA_BF16(a0, b2, in0);  in1 = WMMA_BF16(a1, b2, in1);
  }
  // hidden-to-hidden gates: K = 1024
  for (int k0 = 0; k0 < H_D; k0 += 32) {
    v16bf a0 = load_a_bf16(Ahh, m0,      H_D, k0, lane);
    v16bf a1 = load_a_bf16(Ahh, m0 + 16, H_D, k0, lane);
    v16bf b0 = load_b_bf16(Whh, n0,           H_D, k0, lane);
    v16bf b1 = load_b_bf16(Whh, H_D + n0,     H_D, k0, lane);
    v16bf b2 = load_b_bf16(Whh, 2 * H_D + n0, H_D, k0, lane);
    hr0 = WMMA_BF16(a0, b0, hr0);  hr1 = WMMA_BF16(a1, b0, hr1);
    hz0 = WMMA_BF16(a0, b1, hz0);  hz1 = WMMA_BF16(a1, b1, hz1);
    hn0 = WMMA_BF16(a0, b2, hn0);  hn1 = WMMA_BF16(a1, b2, hn1);
  }

  // C/D layout: VGPR e, lanes 0-15 -> M=e, lanes 16-31 -> M=e+8; N = lane&15.
  const int col = n0 + (lane & 15);
  const float bir = bih[col], biz = bih[H_D + col], bin = bih[2 * H_D + col];
  const float bhr = bhh[col], bhz = bhh[H_D + col], bhn = bhh[2 * H_D + col];
#pragma unroll
  for (int half = 0; half < 2; ++half) {
    const v8f& ir = half ? ir1 : ir0;
    const v8f& iz = half ? iz1 : iz0;
    const v8f& in_ = half ? in1 : in0;
    const v8f& hr = half ? hr1 : hr0;
    const v8f& hz = half ? hz1 : hz0;
    const v8f& hn = half ? hn1 : hn0;
    const int mbase = m0 + half * 16;
#pragma unroll
    for (int e = 0; e < 8; ++e) {
      const int row = mbase + e + ((lane >> 4) << 3);
      const float r = 1.0f / (1.0f + expf(-(ir[e] + bir + hr[e] + bhr)));
      const float z = 1.0f / (1.0f + expf(-(iz[e] + biz + hz[e] + bhz)));
      const float n = tanhf(in_[e] + bin + r * (hn[e] + bhn));
      const float hp = Hprev[(size_t)row * H_D + col];
      const float h = (1.0f - z) * n + z * hp;
      Hout[(size_t)row * H_D + col] = h;
      HoutB[(size_t)row * H_D + col] = f2bf(h);
    }
  }
}

// ---------------------------------------------------------------------------
// Head GEMM: Out[B,N] = gelu(A[B,1024] @ W[N,1024]^T + bias), exact gelu (erf).
// ---------------------------------------------------------------------------
__global__ __launch_bounds__(256)
void head_gemm_gelu_kernel(const unsigned short* __restrict__ A, // [B,1024] bf16
                           const unsigned short* __restrict__ W, // [N,1024] bf16
                           const float* __restrict__ bias,
                           float* __restrict__ Out, int N)
{
  const int lane = threadIdx.x & 31;
  const int wave = threadIdx.x >> 5;
  const int tile = blockIdx.x * 8 + wave;
  const int nblks = N >> 4;
  const int nblk = tile % nblks;
  const int mblk = tile / nblks;
  const int m0 = mblk * 16;
  const int n0 = nblk * 16;

  v8f acc = {0,0,0,0,0,0,0,0};
  for (int k0 = 0; k0 < H_D; k0 += 32) {
    v16bf a = load_a_bf16(A, m0, H_D, k0, lane);
    v16bf b = load_b_bf16(W, n0, H_D, k0, lane);
    acc = WMMA_BF16(a, b, acc);
  }
  const int col = n0 + (lane & 15);
  const float bb = bias[col];
#pragma unroll
  for (int e = 0; e < 8; ++e) {
    const int row = m0 + e + ((lane >> 4) << 3);
    const float x = acc[e] + bb;
    Out[(size_t)row * N + col] = 0.5f * x * (1.0f + erff(x * 0.70710678118654752f));
  }
}

// ---------------------------------------------------------------------------
// Conversion / packing kernels
// ---------------------------------------------------------------------------
__global__ void f32_to_bf16_kernel(const float* __restrict__ in,
                                   unsigned short* __restrict__ out, int n) {
  int i = blockIdx.x * blockDim.x + threadIdx.x;
  if (i < n) out[i] = f2bf(in[i]);
}

// w_ih0 [3072,33] -> bf16 [3072,64], zero-padded K
__global__ void pad_wih0_kernel(const float* __restrict__ in,
                                unsigned short* __restrict__ out) {
  int i = blockIdx.x * blockDim.x + threadIdx.x;
  if (i >= TH_D * 64) return;
  int r = i >> 6, k = i & 63;
  out[i] = (k < 33) ? f2bf(in[r * 33 + k]) : (unsigned short)0;
}

// x = concat(neuron_output [B,32], embedding [B,1]) zero-padded to bf16 [B,64]
__global__ void build_xpad_kernel(const float* __restrict__ no,
                                  const float* __restrict__ emb,
                                  unsigned short* __restrict__ xpad) {
  int i = blockIdx.x * blockDim.x + threadIdx.x;
  if (i >= B_N * 64) return;
  int r = i >> 6, k = i & 63;
  float v = 0.0f;
  if (k < 32)       v = no[r * 32 + k];
  else if (k == 32) v = emb[r];
  xpad[i] = f2bf(v);
}

// ---------------------------------------------------------------------------
// Final heads: one wave32 per batch row; shuffle-reduced dots + control logic
// ---------------------------------------------------------------------------
__device__ __forceinline__ float wsum(float v) {
#pragma unroll
  for (int m = 16; m >= 1; m >>= 1) v += __shfl_xor(v, m, 32);
  return v;
}

__global__ __launch_bounds__(256)
void heads_finalize_kernel(const float* __restrict__ gru_out,  // [B,1024] fp32
                           const float* __restrict__ e1,       // [B,512]
                           const float* __restrict__ p1,       // [B,64]
                           const float* __restrict__ s1,       // [B,64]
                           const float* __restrict__ flow_age,
                           const float* __restrict__ ep_w2, const float* __restrict__ ep_b2,
                           const float* __restrict__ pp_w2, const float* __restrict__ pp_b2,
                           const float* __restrict__ sg_w2, const float* __restrict__ sg_b2,
                           const float* __restrict__ se_w,  const float* __restrict__ se_b,
                           float* __restrict__ out)
{
  const int lane = threadIdx.x & 31;
  const int row = (blockIdx.x * blockDim.x + threadIdx.x) >> 5;
  if (row >= B_N) return;

  const float* g = gru_out + (size_t)row * H_D;
  float se = 0.0f;
  for (int k = lane; k < H_D; k += 32) se += g[k] * se_w[k];

  const float* er = e1 + (size_t)row * 512;
  float ed = 0.0f;
  for (int k = lane; k < 512; k += 32) ed += er[k] * ep_w2[k];

  const float* pr = p1 + (size_t)row * 64;
  float d0 = 0.0f, d1 = 0.0f, d2 = 0.0f;
  for (int k = lane; k < 64; k += 32) {
    const float v = pr[k];
    d0 += v * pp_w2[k];
    d1 += v * pp_w2[64 + k];
    d2 += v * pp_w2[128 + k];
  }

  const float* sr = s1 + (size_t)row * 64;
  float sg = 0.0f;
  for (int k = lane; k < 64; k += 32) sg += sr[k] * sg_w2[k];

  se = wsum(se); ed = wsum(ed); d0 = wsum(d0); d1 = wsum(d1); d2 = wsum(d2); sg = wsum(sg);

  if (lane == 0) {
    const float energy = tanhf(ed + ep_b2[0]);
    // scale = [W-1, H-1, 2D-1] = [63,63,63]
    const float px = (tanhf(d0 + pp_b2[0]) + 1.0f) * 0.5f * 63.0f;
    const float py = (tanhf(d1 + pp_b2[1]) + 1.0f) * 0.5f * 63.0f;
    float pz = (tanhf(d2 + pp_b2[2]) + 1.0f) * 0.5f * 63.0f;
    pz += 0.5f + flow_age[row] * 0.1f;

    const bool oob_xy  = (px < 0.0f) || (px >= 64.0f) || (py < 0.0f) || (py >= 64.0f);
    const bool reached = (pz >= 32.0f) && (pz <= 63.0f);
    const bool oob_z   = pz > 63.0f;
    const bool term    = oob_xy || reached || oob_z;

    const float nx = rintf(px);               // round-half-even == jnp.round
    const float ny = rintf(py);
    const float nz = reached ? 32.0f : rintf(pz);

    const float prob = 1.0f / (1.0f + expf(-(sg + sg_b2[0])));
    const bool decided = prob > 0.5f;
    int raw = (int)floorf((prob - 0.5f) * 2.0f * 3.0f) + 1;
    const int counts = decided ? (raw < 3 ? raw : 3) : 0;
    const float sen = tanhf(se + se_b[0]);
    const float frac = decided ? sen / (float)(counts + 1) : 0.0f;

    out[row] = energy;
    out[(size_t)B_N + (size_t)row * 3 + 0] = nx;
    out[(size_t)B_N + (size_t)row * 3 + 1] = ny;
    out[(size_t)B_N + (size_t)row * 3 + 2] = nz;
    out[(size_t)4 * B_N + row] = frac;
    out[(size_t)5 * B_N + row] = (float)counts;
    out[(size_t)6 * B_N + row] = term ? 1.0f : 0.0f;
  }
}

// ---------------------------------------------------------------------------
extern "C" void kernel_launch(void* const* d_in, const int* in_sizes, int n_in,
                              void* d_out, int out_size, void* d_ws, size_t ws_size,
                              hipStream_t stream) {
  const float* neuron_output = (const float*)d_in[0];
  const float* embedding    = (const float*)d_in[1];
  const float* hidden       = (const float*)d_in[2];   // [2,B,H]
  const float* flow_age     = (const float*)d_in[3];
  const float* w_ih0 = (const float*)d_in[4];
  const float* w_hh0 = (const float*)d_in[5];
  const float* b_ih0 = (const float*)d_in[6];
  const float* b_hh0 = (const float*)d_in[7];
  const float* w_ih1 = (const float*)d_in[8];
  const float* w_hh1 = (const float*)d_in[9];
  const float* b_ih1 = (const float*)d_in[10];
  const float* b_hh1 = (const float*)d_in[11];
  const float* ep_w1 = (const float*)d_in[12];
  const float* ep_b1 = (const float*)d_in[13];
  const float* ep_w2 = (const float*)d_in[14];
  const float* ep_b2 = (const float*)d_in[15];
  const float* pp_w1 = (const float*)d_in[16];
  const float* pp_b1 = (const float*)d_in[17];
  const float* pp_w2 = (const float*)d_in[18];
  const float* pp_b2 = (const float*)d_in[19];
  const float* sg_w1 = (const float*)d_in[20];
  const float* sg_b1 = (const float*)d_in[21];
  const float* sg_w2 = (const float*)d_in[22];
  const float* sg_b2 = (const float*)d_in[23];
  const float* se_w  = (const float*)d_in[24];
  const float* se_b  = (const float*)d_in[25];
  float* out = (float*)d_out;

  // workspace carve-out (256B aligned regions)
  char* ws = (char*)d_ws;
  size_t off = 0;
  auto alloc = [&](size_t bytes) -> char* {
    char* p = ws + off;
    off = (off + bytes + 255) & ~(size_t)255;
    return p;
  };
  unsigned short* xpad  = (unsigned short*)alloc((size_t)B_N * 64 * 2);
  unsigned short* wih0b = (unsigned short*)alloc((size_t)TH_D * 64 * 2);
  unsigned short* whh0b = (unsigned short*)alloc((size_t)TH_D * H_D * 2);
  unsigned short* wih1b = (unsigned short*)alloc((size_t)TH_D * H_D * 2);
  unsigned short* whh1b = (unsigned short*)alloc((size_t)TH_D * H_D * 2);
  unsigned short* epw1b = (unsigned short*)alloc((size_t)512 * H_D * 2);
  unsigned short* ppw1b = (unsigned short*)alloc((size_t)64 * H_D * 2);
  unsigned short* sgw1b = (unsigned short*)alloc((size_t)64 * H_D * 2);
  unsigned short* hprevb = (unsigned short*)alloc((size_t)2 * B_N * H_D * 2); // [2,B,H] bf16
  unsigned short* h0b   = (unsigned short*)alloc((size_t)B_N * H_D * 2);
  unsigned short* h1b   = (unsigned short*)alloc((size_t)B_N * H_D * 2);
  float*          e1    = (float*)         alloc((size_t)B_N * 512 * 4);
  float*          p1    = (float*)         alloc((size_t)B_N * 64 * 4);
  float*          s1    = (float*)         alloc((size_t)B_N * 64 * 4);

  // new_hidden region of d_out: after energy(B) + pos(3B) + frac(B) + counts(B) + term(B)
  float* h0_out = out + (size_t)7 * B_N;
  float* h1_out = h0_out + (size_t)B_N * H_D;

  const dim3 blk(256);

  // pack inputs / weights / prev hidden to bf16 (once)
  build_xpad_kernel<<<(B_N * 64 + 255) / 256, blk, 0, stream>>>(neuron_output, embedding, xpad);
  pad_wih0_kernel<<<(TH_D * 64 + 255) / 256, blk, 0, stream>>>(w_ih0, wih0b);
  const int nw = TH_D * H_D;
  f32_to_bf16_kernel<<<(nw + 255) / 256, blk, 0, stream>>>(w_hh0, whh0b, nw);
  f32_to_bf16_kernel<<<(nw + 255) / 256, blk, 0, stream>>>(w_ih1, wih1b, nw);
  f32_to_bf16_kernel<<<(nw + 255) / 256, blk, 0, stream>>>(w_hh1, whh1b, nw);
  f32_to_bf16_kernel<<<(512 * H_D + 255) / 256, blk, 0, stream>>>(ep_w1, epw1b, 512 * H_D);
  f32_to_bf16_kernel<<<(64 * H_D + 255) / 256, blk, 0, stream>>>(pp_w1, ppw1b, 64 * H_D);
  f32_to_bf16_kernel<<<(64 * H_D + 255) / 256, blk, 0, stream>>>(sg_w1, sgw1b, 64 * H_D);
  const int nh = 2 * B_N * H_D;
  f32_to_bf16_kernel<<<(nh + 255) / 256, blk, 0, stream>>>(hidden, hprevb, nh);

  // fused GRU layers (stream-ordered: layer1 consumes layer0's bf16 output)
  const int gru_blocks = (B_N / 32) * (H_D / 16) / 8;   // 8192 blocks of 8 waves
  gru_layer_kernel<<<gru_blocks, blk, 0, stream>>>(
      xpad, 64, wih0b, b_ih0,
      hprevb, whh0b, b_hh0,
      hidden, h0_out, h0b);
  gru_layer_kernel<<<gru_blocks, blk, 0, stream>>>(
      h0b, H_D, wih1b, b_ih1,
      hprevb + (size_t)B_N * H_D, whh1b, b_hh1,
      hidden + (size_t)B_N * H_D, h1_out, h1b);

  // heads (first linear + gelu) via WMMA, bf16 A from h1b
  head_gemm_gelu_kernel<<<(B_N / 16) * (512 / 16) / 8, blk, 0, stream>>>(h1b, epw1b, ep_b1, e1, 512);
  head_gemm_gelu_kernel<<<(B_N / 16) * (64 / 16) / 8, blk, 0, stream>>>(h1b, ppw1b, pp_b1, p1, 64);
  head_gemm_gelu_kernel<<<(B_N / 16) * (64 / 16) / 8, blk, 0, stream>>>(h1b, sgw1b, sg_b1, s1, 64);

  // final small dots + control logic, one wave per row
  heads_finalize_kernel<<<(B_N * 32) / 256, blk, 0, stream>>>(
      h1_out, e1, p1, s1, flow_age,
      ep_w2, ep_b2, pp_w2, pp_b2, sg_w2, sg_b2, se_w, se_b, out);
}
